// Seq2Seq_25958782337450
// MI455X (gfx1250) — compile-verified, matrix operand order checked by reading, not run on hardware
//
#include <hip/hip_runtime.h>
#include <hip/hip_bf16.h>

typedef __attribute__((ext_vector_type(16))) __bf16 v16bf;
typedef __attribute__((ext_vector_type(8)))  __bf16 v8bf;
typedef __attribute__((ext_vector_type(8)))  float  v8f;

#define DEV __device__ __forceinline__

DEV float sigm(float x) { return 1.0f / (1.0f + __expf(-x)); }

DEV v8f wmma_bf16(v16bf a, v16bf b, v8f c) {
  return __builtin_amdgcn_wmma_f32_16x16x32_bf16(false, a, false, b, (short)0, c,
                                                 false, false);
}

// Load one 16x32 bf16 operand tile (A or B) in the CDNA5 WMMA register layout.
// `base` points at element (tile_row0, 0) of a row-major matrix, leading dim `ld`.
//   lanes 0-15 : elems 0..7 = K k..k+7,   elems 8..15 = K k+16..k+23
//   lanes 16-31: elems 0..7 = K k+8..k+15, elems 8..15 = K k+24..k+31
DEV v16bf load_tile_ab(const __bf16* base, int ld, int k0, int lane) {
  int r  = lane & 15;
  int hi = lane >> 4;
  const __bf16* p = base + (size_t)r * ld + k0 + hi * 8;
  v8bf lo = *(const v8bf*)(p);
  v8bf hh = *(const v8bf*)(p + 16);
  v16bf out;
#pragma unroll
  for (int i = 0; i < 8; ++i) { out[i] = lo[i]; out[8 + i] = hh[i]; }
  return out;
}

// -------- elementwise helpers --------
__global__ void cvt_f32_bf16(const float* __restrict__ s, __bf16* __restrict__ d,
                             long long n) {
  long long i = (long long)blockIdx.x * blockDim.x + threadIdx.x;
  if (i < n) d[i] = (__bf16)s[i];
}

__global__ void fill_zero_f32(float* __restrict__ p, long long n) {
  long long i = (long long)blockIdx.x * blockDim.x + threadIdx.x;
  if (i < n) p[i] = 0.0f;
}

__global__ void embed_bf16(const int* __restrict__ idx, const float* __restrict__ emb,
                           __bf16* __restrict__ out, long long n, int E) {
  long long i = (long long)blockIdx.x * blockDim.x + threadIdx.x;
  if (i >= n) return;
  long long tok = i / E;
  int e = (int)(i % E);
  out[i] = (__bf16)emb[(size_t)idx[tok] * E + e];
}

// -------- GEMM: Out[M,N] = X[M,K] * W[N,K]^T + bias[N]  (bf16 in, f32 out) --------
// 256 threads = 8 waves. Each wave: 2 M-tiles x 4 N-tiles (8 WMMAs / K-step).
// K-loop is unrolled by 2 with an explicit ping-pong buffer index so the
// double-buffer "rotation" is pure register renaming (no v_mov copies, fewer
// WMMA->VALU hazard NOPs), while loads for step k+1 overlap WMMAs of step k.
__global__ __launch_bounds__(256)
void gemm_bias_bf16(const __bf16* __restrict__ X, const __bf16* __restrict__ W,
                    const float* __restrict__ bias, float* __restrict__ Out,
                    int M, int N, int K, int Ntiles) {
  int wave = threadIdx.x >> 5;
  int lane = threadIdx.x & 31;
  int lane15 = lane & 15;
  int hi = lane >> 4;
  int row0 = blockIdx.x * 32;                 // two M-tiles per wave
  int nt0 = (blockIdx.y * 8 + wave) * 4;

  const __bf16* Xa = X + (size_t)row0 * K;
  const __bf16* Xb = X + (size_t)(row0 + 16) * K;

  int nt[4];
  v8f acc0[4], acc1[4];
#pragma unroll
  for (int j = 0; j < 4; ++j) {
    int t = nt0 + j;
    if (t > Ntiles - 1) t = Ntiles - 1;       // clamp: keeps EXEC all-ones at WMMA
    nt[j] = t;
    float bv = bias[t * 16 + lane15];
#pragma unroll
    for (int r = 0; r < 8; ++r) { acc0[j][r] = bv; acc1[j][r] = bv; }
  }

  const int KS = K >> 5;  // always even here (K = 512)
  v16bf A0[2], A1[2], Bt[2][4];
  A0[0] = load_tile_ab(Xa, K, 0, lane);
  A1[0] = load_tile_ab(Xb, K, 0, lane);
#pragma unroll
  for (int j = 0; j < 4; ++j)
    Bt[0][j] = load_tile_ab(W + (size_t)nt[j] * 16 * K, K, 0, lane);

#pragma unroll 2
  for (int ks = 0; ks < KS; ++ks) {
    const int pp = ks & 1;        // compile-time constant in each unrolled body
    const int np = pp ^ 1;
    int kn = (ks + 1 < KS) ? (ks + 1) * 32 : ks * 32;  // clamped redundant tail load
    A0[np] = load_tile_ab(Xa, K, kn, lane);
    A1[np] = load_tile_ab(Xb, K, kn, lane);
#pragma unroll
    for (int j = 0; j < 4; ++j)
      Bt[np][j] = load_tile_ab(W + (size_t)nt[j] * 16 * K, K, kn, lane);
#pragma unroll
    for (int j = 0; j < 4; ++j) {
      acc0[j] = wmma_bf16(A0[pp], Bt[pp][j], acc0[j]);
      acc1[j] = wmma_bf16(A1[pp], Bt[pp][j], acc1[j]);
    }
  }

#pragma unroll
  for (int j = 0; j < 4; ++j) {
    if (nt0 + j < Ntiles) {
      int n0 = nt[j] * 16;
#pragma unroll
      for (int r = 0; r < 8; ++r) {
        Out[(size_t)(row0 + hi * 8 + r) * N + n0 + lane15] = acc0[j][r];
        Out[(size_t)(row0 + 16 + hi * 8 + r) * N + n0 + lane15] = acc1[j][r];
      }
    }
  }
}

// -------- LSTM scan over T steps, one persistent workgroup (32 wave32) --------
// xW : [B*T, 4H] f32 (row = b*T + t), gate order i,f,g,o along 4H
// Whh: [4H, H] bf16 row-major; h kept in LDS (bf16), c in registers (f32, C-layout)
__global__ __launch_bounds__(1024)
void lstm_scan(const float* __restrict__ xW, const __bf16* __restrict__ Whh,
               const float* __restrict__ h0, const float* __restrict__ c0,
               __bf16* __restrict__ hseq,  // [B, T, H]
               float* __restrict__ hT, float* __restrict__ cT, int T) {
  const int H = 512, G4 = 2048;
  __shared__ __bf16 hbuf[16 * 512];  // 16KB: h_t as bf16, row-major [16][512]
  int tid = threadIdx.x;
  int wave = tid >> 5, lane = tid & 31;
  int lane15 = lane & 15, hi = lane >> 4;
  int ncol0 = wave * 16;  // this wave owns h columns [ncol0, ncol0+16)

  for (int i = tid; i < 16 * H; i += 1024) hbuf[i] = (__bf16)h0[i];

  v8f cst, hlast;
#pragma unroll
  for (int r = 0; r < 8; ++r) {
    cst[r] = c0[(size_t)(hi * 8 + r) * H + ncol0 + lane15];
    hlast[r] = 0.0f;
  }
  __syncthreads();

  for (int t = 0; t < T; ++t) {
    // warm L1 for next step's accumulator init while this step computes
    if (t + 1 < T) {
#pragma unroll
      for (int g = 0; g < 4; ++g)
        __builtin_prefetch(
            &xW[(size_t)((hi * 8) * T + t + 1) * G4 + g * H + ncol0 + lane15], 0, 3);
    }
    // accumulators start from the hoisted xW (bias already folded in)
    v8f acc[4];
#pragma unroll
    for (int g = 0; g < 4; ++g) {
#pragma unroll
      for (int r = 0; r < 8; ++r)
        acc[g][r] = xW[(size_t)((hi * 8 + r) * T + t) * G4 + g * H + ncol0 + lane15];
    }
    // g += h_{t-1} @ Whh^T : issue all 4 gate B-tiles as one clause, then
    // 4 back-to-back WMMAs (one wait per K-step instead of four)
    for (int k = 0; k < H; k += 32) {
      v16bf a = load_tile_ab(hbuf, H, k, lane);
      v16bf b0 = load_tile_ab(Whh + (size_t)(0 * H + ncol0) * H, H, k, lane);
      v16bf b1 = load_tile_ab(Whh + (size_t)(1 * H + ncol0) * H, H, k, lane);
      v16bf b2 = load_tile_ab(Whh + (size_t)(2 * H + ncol0) * H, H, k, lane);
      v16bf b3 = load_tile_ab(Whh + (size_t)(3 * H + ncol0) * H, H, k, lane);
      acc[0] = wmma_bf16(a, b0, acc[0]);
      acc[1] = wmma_bf16(a, b1, acc[1]);
      acc[2] = wmma_bf16(a, b2, acc[2]);
      acc[3] = wmma_bf16(a, b3, acc[3]);
    }
    __syncthreads();  // everyone done reading h_{t-1} from LDS
#pragma unroll
    for (int r = 0; r < 8; ++r) {
      float iv = sigm(acc[0][r]);
      float fv = sigm(acc[1][r]);
      float gv = tanhf(acc[2][r]);
      float ov = sigm(acc[3][r]);
      float c = fv * cst[r] + iv * gv;
      cst[r] = c;
      hlast[r] = ov * tanhf(c);
      __bf16 hb = (__bf16)hlast[r];
      int m = hi * 8 + r;  // batch index
      hbuf[m * H + ncol0 + lane15] = hb;
      hseq[(size_t)(m * T + t) * H + ncol0 + lane15] = hb;
    }
    __syncthreads();  // h_t visible before next step's A loads
  }
#pragma unroll
  for (int r = 0; r < 8; ++r) {
    int m = hi * 8 + r;
    hT[(size_t)m * H + ncol0 + lane15] = hlast[r];
    cT[(size_t)m * H + ncol0 + lane15] = cst[r];
  }
}

extern "C" void kernel_launch(void* const* d_in, const int* in_sizes, int n_in,
                              void* d_out, int out_size, void* d_ws, size_t ws_size,
                              hipStream_t stream) {
  constexpr int B = 16, S = 128, T = 128, E = 512, H = 512, NL = 4, V = 32000;
  constexpr int G4 = 4 * H;  // 2048

  const int*   src  = (const int*)d_in[0];
  const int*   tgt  = (const int*)d_in[1];
  const float* semb = (const float*)d_in[2];
  const float* temb = (const float*)d_in[3];
  const float* eWih = (const float*)d_in[4];
  const float* eWhh = (const float*)d_in[5];
  const float* eb   = (const float*)d_in[6];
  const float* dWih = (const float*)d_in[7];
  const float* dWhh = (const float*)d_in[8];
  const float* db   = (const float*)d_in[9];
  const float* fcW  = (const float*)d_in[10];
  const float* fcb  = (const float*)d_in[11];
  float* out = (float*)d_out;

  char* ws = (char*)d_ws;
  size_t off = 0;
  auto take = [&](size_t bytes) -> void* {
    void* p = ws + off;
    off = (off + bytes + 255) & ~(size_t)255;
    return p;
  };
  __bf16* eWih_b = (__bf16*)take((size_t)NL * G4 * E * 2);
  __bf16* eWhh_b = (__bf16*)take((size_t)NL * G4 * H * 2);
  __bf16* dWih_b = (__bf16*)take((size_t)NL * G4 * E * 2);
  __bf16* dWhh_b = (__bf16*)take((size_t)NL * G4 * H * 2);
  __bf16* fcW_b  = (__bf16*)take((size_t)V * H * 2);
  __bf16* xa     = (__bf16*)take((size_t)B * T * H * 2);
  __bf16* xb     = (__bf16*)take((size_t)B * T * H * 2);
  float*  xWbuf  = (float*)take((size_t)B * T * G4 * 4);
  float*  zeroS  = (float*)take((size_t)B * H * 4);
  float*  hN     = (float*)take((size_t)NL * B * H * 4);
  float*  cN     = (float*)take((size_t)NL * B * H * 4);
  float*  hDump  = (float*)take((size_t)B * H * 4);
  float*  cDump  = (float*)take((size_t)B * H * 4);

  auto cvt = [&](const float* s, __bf16* d, long long n) {
    cvt_f32_bf16<<<dim3((unsigned)((n + 255) / 256)), dim3(256), 0, stream>>>(s, d, n);
  };

  fill_zero_f32<<<dim3((B * H + 255) / 256), dim3(256), 0, stream>>>(zeroS, (long long)B * H);
  cvt(eWih, eWih_b, (long long)NL * G4 * E);
  cvt(eWhh, eWhh_b, (long long)NL * G4 * H);
  cvt(dWih, dWih_b, (long long)NL * G4 * E);
  cvt(dWhh, dWhh_b, (long long)NL * G4 * H);
  cvt(fcW,  fcW_b,  (long long)V * H);

  long long nx = (long long)B * S * E;
  embed_bf16<<<dim3((unsigned)((nx + 255) / 256)), dim3(256), 0, stream>>>(src, semb, xa, nx, E);

  __bf16* cur = xa;
  __bf16* nxt = xb;

  // ---- encoder ----
  for (int l = 0; l < NL; ++l) {
    gemm_bias_bf16<<<dim3(B * T / 32, (G4 / 16) / 32), dim3(256), 0, stream>>>(
        cur, eWih_b + (size_t)l * G4 * E, eb + (size_t)l * G4, xWbuf,
        B * T, G4, E, G4 / 16);
    lstm_scan<<<dim3(1), dim3(1024), 0, stream>>>(
        xWbuf, eWhh_b + (size_t)l * G4 * H, zeroS, zeroS, nxt,
        hN + (size_t)l * B * H, cN + (size_t)l * B * H, T);
    __bf16* tmp = cur; cur = nxt; nxt = tmp;
  }

  // ---- decoder (init from encoder final states) ----
  embed_bf16<<<dim3((unsigned)((nx + 255) / 256)), dim3(256), 0, stream>>>(tgt, temb, cur, nx, E);
  for (int l = 0; l < NL; ++l) {
    gemm_bias_bf16<<<dim3(B * T / 32, (G4 / 16) / 32), dim3(256), 0, stream>>>(
        cur, dWih_b + (size_t)l * G4 * E, db + (size_t)l * G4, xWbuf,
        B * T, G4, E, G4 / 16);
    lstm_scan<<<dim3(1), dim3(1024), 0, stream>>>(
        xWbuf, dWhh_b + (size_t)l * G4 * H, hN + (size_t)l * B * H, cN + (size_t)l * B * H,
        nxt, hDump, cDump, T);
    __bf16* tmp = cur; cur = nxt; nxt = tmp;
  }

  // ---- vocab projection: [2048,512] x [32000,512]^T + fc_b -> d_out ----
  int Ntiles = V / 16;  // 2000
  gemm_bias_bf16<<<dim3(B * T / 32, (Ntiles + 31) / 32), dim3(256), 0, stream>>>(
      cur, fcW_b, fcb, out, B * T, V, H, Ntiles);

  (void)in_sizes; (void)n_in; (void)out_size; (void)ws_size;
}
